// USVHeteroGNN_68848325755098
// MI455X (gfx1250) — compile-verified, matrix operand order checked by reading, not run on hardware
//
#include <hip/hip_runtime.h>
#include <stdint.h>

// ---------------------------------------------------------------- types
typedef __attribute__((ext_vector_type(16))) __bf16 v16bf;
typedef __attribute__((ext_vector_type(8)))  float  v8f;

// ---------------------------------------------------------------- device helpers
static __device__ __forceinline__ unsigned short f2bf(float f) {
    unsigned u = __float_as_uint(f);
    u += 0x7FFFu + ((u >> 16) & 1u);           // round-to-nearest-even
    return (unsigned short)(u >> 16);
}
static __device__ __forceinline__ unsigned fkey(float f) {   // order-preserving f32->u32
    unsigned u = __float_as_uint(f);
    return (u & 0x80000000u) ? ~u : (u | 0x80000000u);
}
static __device__ __forceinline__ float fdecode(unsigned k) {
    unsigned u = (k & 0x80000000u) ? (k & 0x7FFFFFFFu) : ~k;
    return __uint_as_float(u);
}
static __device__ __forceinline__ float lrelu(float v) { return v > 0.f ? v : 0.2f * v; }

// ---------------------------------------------------------------- kernels
__global__ void fill0_k(unsigned* __restrict__ p, int n) {
    int i = blockIdx.x * blockDim.x + threadIdx.x;
    if (i < n) p[i] = 0u;
}

// Pack f32 activation matrix [N,K] into bf16 WMMA A-fragments.
// Fragment f = rowtile*Kt + ktile; per fragment: 32 lanes x 16 bf16 (v16bf/lane).
// A-layout (16-bit 16x32): lane = m + 16*g ; elem e -> k = (e<8 ? g*8+e : 16+g*8+e-8)
__global__ void pack_a_k(const float* __restrict__ x, unsigned short* __restrict__ dst,
                         int N, int K, int Kt) {
    int idx  = blockIdx.x * blockDim.x + threadIdx.x;
    int lane = idx & 31, f = idx >> 5;
    int rowtiles = (N + 15) >> 4;
    if (f >= rowtiles * Kt) return;
    int kt = f % Kt, rt = f / Kt;
    int m = lane & 15, g = lane >> 4;
    int row = rt * 16 + m;
    unsigned short* o = dst + (size_t)f * 512 + lane * 16;
#pragma unroll
    for (int e = 0; e < 16; ++e) {
        int koff = (e < 8) ? (g * 8 + e) : (16 + g * 8 + (e - 8));
        int k = kt * 32 + koff;
        float v = (row < N && k < K) ? x[(size_t)row * K + k] : 0.f;
        o[e] = f2bf(v);
    }
}

// Pack f32 weight matrix [K,Nout] into bf16 B-fragments (B^T uses the A layout with N as M).
__global__ void pack_w_k(const float* __restrict__ w, unsigned short* __restrict__ dst,
                         int K, int Nout, int Kt, int coltiles) {
    int idx  = blockIdx.x * blockDim.x + threadIdx.x;
    int lane = idx & 31, f = idx >> 5;
    if (f >= coltiles * Kt) return;
    int kt = f % Kt, ct = f / Kt;
    int nl = lane & 15, g = lane >> 4;
    int n = ct * 16 + nl;
    unsigned short* o = dst + (size_t)f * 512 + lane * 16;
#pragma unroll
    for (int e = 0; e < 16; ++e) {
        int koff = (e < 8) ? (g * 8 + e) : (16 + g * 8 + (e - 8));
        int k = kt * 32 + koff;
        float v = (k < K && n < Nout) ? w[(size_t)k * Nout + n] : 0.f;
        o[e] = f2bf(v);
    }
}

// One wave per 16x64 C stripe (4 col tiles): each A fragment feeds 4 WMMAs from registers.
__global__ __launch_bounds__(32) void wmma_gemm_k(
    const unsigned short* __restrict__ Ap, const unsigned short* __restrict__ Wp,
    const float* __restrict__ bias, float* __restrict__ out,
    int N, int Nout, int Kt, int relu) {
    int rt = blockIdx.x, ct0 = blockIdx.y * 4, lane = threadIdx.x;
    const v16bf* A = (const v16bf*)Ap + (size_t)rt * Kt * 32 + lane;
    const v16bf* B = (const v16bf*)Wp + lane;
    v8f acc0 = {0.f,0.f,0.f,0.f,0.f,0.f,0.f,0.f};
    v8f acc1 = acc0, acc2 = acc0, acc3 = acc0;
    for (int kt = 0; kt < Kt; ++kt) {
        v16bf a  = A[(size_t)kt * 32];
        __builtin_prefetch(&A[(size_t)(kt + 1) * 32], 0, 1);   // global_prefetch_b8
        v16bf b0 = B[((size_t)(ct0 + 0) * Kt + kt) * 32];
        v16bf b1 = B[((size_t)(ct0 + 1) * Kt + kt) * 32];
        v16bf b2 = B[((size_t)(ct0 + 2) * Kt + kt) * 32];
        v16bf b3 = B[((size_t)(ct0 + 3) * Kt + kt) * 32];
        acc0 = __builtin_amdgcn_wmma_f32_16x16x32_bf16(false, a, false, b0, (short)0, acc0, false, false);
        acc1 = __builtin_amdgcn_wmma_f32_16x16x32_bf16(false, a, false, b1, (short)0, acc1, false, false);
        acc2 = __builtin_amdgcn_wmma_f32_16x16x32_bf16(false, a, false, b2, (short)0, acc2, false, false);
        acc3 = __builtin_amdgcn_wmma_f32_16x16x32_bf16(false, a, false, b3, (short)0, acc3, false, false);
    }
    // C layout: lanes 0-15 -> N=lane, M=v ; lanes 16-31 -> N=lane-16, M=v+8
    int nl = lane & 15, g = lane >> 4;
    v8f accs[4] = {acc0, acc1, acc2, acc3};
#pragma unroll
    for (int t = 0; t < 4; ++t) {
        int col = (ct0 + t) * 16 + nl;
        float bv = bias ? bias[col] : 0.f;
#pragma unroll
        for (int v = 0; v < 8; ++v) {
            int row = rt * 16 + v + 8 * g;
            if (row < N) {
                float x = accs[t][v] + bv;
                if (relu) x = fmaxf(x, 0.f);
                out[(size_t)row * Nout + col] = x;
            }
        }
    }
}

// wa[d,h] = sum_c w[d, h*C + c] * a[h,c]
__global__ void reduce_aw_k(const float* __restrict__ w, const float* __restrict__ a,
                            float* __restrict__ wa, int Din, int H, int Cc) {
    int i = blockIdx.x * blockDim.x + threadIdx.x;
    if (i >= Din * H) return;
    int d = i / H, h = i - d * H;
    int HC = H * Cc;
    float acc = 0.f;
    for (int c = 0; c < Cc; ++c) acc += w[(size_t)d * HC + h * Cc + c] * a[h * Cc + c];
    wa[i] = acc;
}

// s[n,h] = x[n,:] . wa[:,h]
__global__ void attn_k(const float* __restrict__ x, const float* __restrict__ wa,
                       float* __restrict__ s, int N, int Din, int H) {
    int i = blockIdx.x * blockDim.x + threadIdx.x;
    if (i >= N * H) return;
    int n = i / H, h = i - n * H;
    const float* xr = x + (size_t)n * Din;
    float acc = 0.f;
    for (int d = 0; d < Din; ++d) acc += xr[d] * wa[d * H + h];
    s[i] = acc;
}

__global__ void edge_max_k(const int* __restrict__ src, const int* __restrict__ dst,
                           int E, int H, const float* __restrict__ as,
                           const float* __restrict__ ad, unsigned* __restrict__ mkey) {
    int i = blockIdx.x * blockDim.x + threadIdx.x;
    if (i >= E * H) return;
    int e = i / H, h = i - e * H;
    float v = lrelu(as[src[e] * H + h] + ad[dst[e] * H + h]);
    atomicMax(&mkey[dst[e] * H + h], fkey(v));
}

__global__ void edge_expsum_k(const int* __restrict__ src, const int* __restrict__ dst,
                              int E, int H, const float* __restrict__ as,
                              const float* __restrict__ ad, const unsigned* __restrict__ mkey,
                              float* __restrict__ alpha, float* __restrict__ den) {
    int i = blockIdx.x * blockDim.x + threadIdx.x;
    if (i >= E * H) return;
    int e = i / H, h = i - e * H;
    float v = lrelu(as[src[e] * H + h] + ad[dst[e] * H + h]);
    float m = fdecode(mkey[dst[e] * H + h]);
    float ex = __expf(v - m);
    alpha[i] = ex;
    atomicAdd(&den[dst[e] * H + h], ex);
}

__global__ void edge_norm_k(const int* __restrict__ dst, int E, int H,
                            float* __restrict__ alpha, const float* __restrict__ den) {
    int i = blockIdx.x * blockDim.x + threadIdx.x;
    if (i >= E * H) return;
    int e = i / H, h = i - e * H;
    alpha[i] = alpha[i] / (den[dst[e] * H + h] + 1e-16f);
}

__global__ void scatter_k(const int* __restrict__ src, const int* __restrict__ dst,
                          int E, int H, int Cc, const float* __restrict__ hs,
                          const float* __restrict__ alpha, float* __restrict__ out) {
    int i = blockIdx.x * blockDim.x + threadIdx.x;
    int HC = H * Cc;
    if (i >= E * HC) return;
    int e = i / HC, r = i - e * HC, h = r / Cc;
    atomicAdd(&out[(size_t)dst[e] * HC + r], hs[(size_t)src[e] * HC + r] * alpha[e * H + h]);
}

// out = relu(in + b1 + b2?)  (may alias in)
__global__ void combine_k(const float* __restrict__ in, const float* __restrict__ b1,
                          const float* __restrict__ b2, float* __restrict__ out,
                          int N, int HC) {
    int i = blockIdx.x * blockDim.x + threadIdx.x;
    if (i >= N * HC) return;
    int c = i % HC;
    float v = in[i] + b1[c] + (b2 ? b2[c] : 0.f);
    out[i] = fmaxf(v, 0.f);
}

__global__ void build_loops_k(const int* __restrict__ s, const int* __restrict__ d,
                              int E, int Nloop, int* __restrict__ so, int* __restrict__ dd) {
    int i = blockIdx.x * blockDim.x + threadIdx.x;
    if (i < E) { so[i] = s[i]; dd[i] = d[i]; }
    else if (i < E + Nloop) { so[i] = i - E; dd[i] = i - E; }
}

__global__ void colmean_k(const float* __restrict__ x, int N, int Cc, float* __restrict__ out) {
    __shared__ float sm[256];
    int c = blockIdx.x;
    float acc = 0.f;
    for (int r = threadIdx.x; r < N; r += blockDim.x) acc += x[(size_t)r * Cc + c];
    sm[threadIdx.x] = acc;
    __syncthreads();
    for (int st = 128; st > 0; st >>= 1) {
        if ((int)threadIdx.x < st) sm[threadIdx.x] += sm[threadIdx.x + st];
        __syncthreads();
    }
    if (threadIdx.x == 0) out[c] = sm[0] / (float)N;
}

__global__ void headlin_k(const float* __restrict__ m, const float* __restrict__ w,
                          const float* __restrict__ b, float* __restrict__ out) {
    int j = threadIdx.x;   // 64 threads
    float acc = b[j];
    for (int d = 0; d < 64; ++d) acc += m[d] * w[d * 64 + j];
    out[j] = fmaxf(acc, 0.f);
}

// ---------------------------------------------------------------- host
static inline int cdiv(long a, int b) { return (int)((a + b - 1) / b); }

extern "C" void kernel_launch(void* const* d_in, const int* in_sizes, int n_in,
                              void* d_out, int out_size, void* d_ws, size_t ws_size,
                              hipStream_t stream) {
    (void)in_sizes; (void)out_size; (void)ws_size;
    if (n_in < 46) return;

    const int NU = 20000, NT = 80000, C = 64, H1 = 4;
    const int E_UT = 200000, E_TT = 200000, E_UU = 100000, E_UUL = E_UU + NU;

    const float* usv_x  = (const float*)d_in[0];
    const float* task_x = (const float*)d_in[1];
    const int* src_ut = (const int*)d_in[2]; const int* dst_ut = (const int*)d_in[3];
    const int* src_tt = (const int*)d_in[4]; const int* dst_tt = (const int*)d_in[5];
    const int* src_uu = (const int*)d_in[6]; const int* dst_uu = (const int*)d_in[7];
    const float* wlin_u = (const float*)d_in[8];  const float* blin_u = (const float*)d_in[9];
    const float* wlin_t = (const float*)d_in[10]; const float* blin_t = (const float*)d_in[11];
    // GAT param blocks: r = 0..5 -> l1.ut, l1.tt, l1.uu, l2.ut, l2.tt, l2.uu
    auto gws = [&](int r){ return (const float*)d_in[12 + 5*r + 0]; };
    auto gwd = [&](int r){ return (const float*)d_in[12 + 5*r + 1]; };
    auto gas = [&](int r){ return (const float*)d_in[12 + 5*r + 2]; };
    auto gad = [&](int r){ return (const float*)d_in[12 + 5*r + 3]; };
    auto gb  = [&](int r){ return (const float*)d_in[12 + 5*r + 4]; };
    const float* gu_w = (const float*)d_in[42]; const float* gu_b = (const float*)d_in[43];
    const float* gt_w = (const float*)d_in[44]; const float* gt_b = (const float*)d_in[45];

    // ---- bump allocator over d_ws
    size_t off = 0;
    auto alloc = [&](size_t bytes) -> char* {
        off = (off + 255) & ~(size_t)255;
        char* p = (char*)d_ws + off;
        off += bytes;
        return p;
    };
    float* XU0   = (float*)alloc((size_t)NU * C * 4);
    float* XT0   = (float*)alloc((size_t)NT * C * 4);
    unsigned short* PXU_in = (unsigned short*)alloc((size_t)1250 * 1 * 512 * 2);
    unsigned short* PXT_in = (unsigned short*)alloc((size_t)5000 * 1 * 512 * 2);
    unsigned short* PXU0   = (unsigned short*)alloc((size_t)1250 * 2 * 512 * 2);
    unsigned short* PXT0   = (unsigned short*)alloc((size_t)5000 * 2 * 512 * 2);
    unsigned short* PXU1   = (unsigned short*)alloc((size_t)1250 * 8 * 512 * 2);
    unsigned short* PXT1   = (unsigned short*)alloc((size_t)5000 * 8 * 512 * 2);
    float* HS    = (float*)alloc((size_t)NT * 256 * 4);
    float* OUT_T = (float*)alloc((size_t)NT * 256 * 4);
    float* OUT_U = (float*)alloc((size_t)NU * 256 * 4);
    float* O2T   = (float*)alloc((size_t)NT * 64 * 4);
    float* O2U   = (float*)alloc((size_t)NU * 64 * 4);
    float* ALPHA = (float*)alloc((size_t)800000 * 4);
    float* ASRC  = (float*)alloc((size_t)320000 * 4);
    float* ADST  = (float*)alloc((size_t)320000 * 4);
    unsigned* MKEY = (unsigned*)alloc((size_t)320000 * 4);
    float* DEN   = (float*)alloc((size_t)320000 * 4);
    float* WAS   = (float*)alloc(1024 * 4);
    float* WAD   = (float*)alloc(1024 * 4);
    unsigned short* PW = (unsigned short*)alloc(64 * 1024);
    int* SUU = (int*)alloc((size_t)E_UUL * 4);
    int* DUU = (int*)alloc((size_t)E_UUL * 4);
    float* MU = (float*)alloc(64 * 4);
    float* MT = (float*)alloc(64 * 4);

    auto zero = [&](void* p, long words) {
        fill0_k<<<cdiv(words, 256), 256, 0, stream>>>((unsigned*)p, (int)words);
    };
    auto gemm = [&](const unsigned short* Ap, const float* W, const float* bias, float* out,
                    int N, int K, int Nout, int relu) {
        int Kt = (K + 31) / 32, ctiles = Nout / 16, rtiles = (N + 15) / 16;
        pack_w_k<<<cdiv((long)ctiles * Kt * 32, 256), 256, 0, stream>>>(W, PW, K, Nout, Kt, ctiles);
        dim3 g(rtiles, ctiles / 4);   // each wave covers 4 col tiles (Nout % 64 == 0)
        wmma_gemm_k<<<g, 32, 0, stream>>>(Ap, PW, bias, out, N, Nout, Kt, relu);
    };
    auto packA = [&](const float* x, unsigned short* dst, int N, int K) {
        int Kt = (K + 31) / 32, rtiles = (N + 15) / 16;
        pack_a_k<<<cdiv((long)rtiles * Kt * 32, 256), 256, 0, stream>>>(x, dst, N, K, Kt);
    };
    // One full GAT relation: softmax-attention aggregation into outAcc (+=).
    auto run_gat = [&](const unsigned short* PXs, const float* Xs, const float* Xd,
                       int Ns, int Nd, int Din, const int* src, const int* dst, int E,
                       int r, int H, float* outAcc) {
        int HC = H * C;
        gemm(PXs, gws(r), nullptr, HS, Ns, Din, HC, 0);                       // hs = Xs @ ws
        reduce_aw_k<<<cdiv((long)Din * H, 128), 128, 0, stream>>>(gws(r), gas(r), WAS, Din, H, C);
        reduce_aw_k<<<cdiv((long)Din * H, 128), 128, 0, stream>>>(gwd(r), gad(r), WAD, Din, H, C);
        attn_k<<<cdiv((long)Ns * H, 256), 256, 0, stream>>>(Xs, WAS, ASRC, Ns, Din, H);
        attn_k<<<cdiv((long)Nd * H, 256), 256, 0, stream>>>(Xd, WAD, ADST, Nd, Din, H);
        zero(MKEY, (long)Nd * H);
        zero(DEN,  (long)Nd * H);
        long EH = (long)E * H;
        edge_max_k   <<<cdiv(EH, 256), 256, 0, stream>>>(src, dst, E, H, ASRC, ADST, MKEY);
        edge_expsum_k<<<cdiv(EH, 256), 256, 0, stream>>>(src, dst, E, H, ASRC, ADST, MKEY, ALPHA, DEN);
        edge_norm_k  <<<cdiv(EH, 256), 256, 0, stream>>>(dst, E, H, ALPHA, DEN);
        scatter_k<<<cdiv((long)E * HC, 256), 256, 0, stream>>>(src, dst, E, H, C, HS, ALPHA, outAcc);
    };

    // ---- input linears (K=16 padded to one 32-wide ktile), ReLU
    packA(usv_x, PXU_in, NU, 16);
    gemm(PXU_in, wlin_u, blin_u, XU0, NU, 16, 64, 1);
    packA(task_x, PXT_in, NT, 16);
    gemm(PXT_in, wlin_t, blin_t, XT0, NT, 16, 64, 1);

    packA(XU0, PXU0, NU, 64);
    packA(XT0, PXT0, NT, 64);

    build_loops_k<<<cdiv(E_UUL, 256), 256, 0, stream>>>(src_uu, dst_uu, E_UU, NU, SUU, DUU);

    // ---- layer 1 (heads=4, concat -> 256); HeteroConv sums ut+tt into OUT_T
    zero(OUT_T, (long)NT * 256);
    zero(OUT_U, (long)NU * 256);
    run_gat(PXU0, XU0, XT0, NU, NT, 64, src_ut, dst_ut, E_UT,  0, H1, OUT_T);
    run_gat(PXT0, XT0, XT0, NT, NT, 64, src_tt, dst_tt, E_TT,  1, H1, OUT_T);
    run_gat(PXU0, XU0, XU0, NU, NU, 64, SUU,    DUU,    E_UUL, 2, H1, OUT_U);
    combine_k<<<cdiv((long)NT * 256, 256), 256, 0, stream>>>(OUT_T, gb(0), gb(1), OUT_T, NT, 256);
    combine_k<<<cdiv((long)NU * 256, 256), 256, 0, stream>>>(OUT_U, gb(2), nullptr, OUT_U, NU, 256);

    packA(OUT_U, PXU1, NU, 256);
    packA(OUT_T, PXT1, NT, 256);

    // ---- layer 2 (heads=1 -> 64)
    zero(O2T, (long)NT * 64);
    zero(O2U, (long)NU * 64);
    run_gat(PXU1, OUT_U, OUT_T, NU, NT, 256, src_ut, dst_ut, E_UT,  3, 1, O2T);
    run_gat(PXT1, OUT_T, OUT_T, NT, NT, 256, src_tt, dst_tt, E_TT,  4, 1, O2T);
    run_gat(PXU1, OUT_U, OUT_U, NU, NU, 256, SUU,    DUU,    E_UUL, 5, 1, O2U);

    // ---- final: relu outputs into d_out, then pooled head linears
    float* xu_out = (float*)d_out;
    float* xt_out = xu_out + (size_t)NU * 64;
    float* gs     = xt_out + (size_t)NT * 64;
    combine_k<<<cdiv((long)NU * 64, 256), 256, 0, stream>>>(O2U, gb(5), nullptr, xu_out, NU, 64);
    combine_k<<<cdiv((long)NT * 64, 256), 256, 0, stream>>>(O2T, gb(3), gb(4),   xt_out, NT, 64);
    colmean_k<<<64, 256, 0, stream>>>(xu_out, NU, 64, MU);
    colmean_k<<<64, 256, 0, stream>>>(xt_out, NT, 64, MT);
    headlin_k<<<1, 64, 0, stream>>>(MU, gu_w, gu_b, gs);
    headlin_k<<<1, 64, 0, stream>>>(MT, gt_w, gt_b, gs + 64);
}
